// DimeNet_45191645889270
// MI455X (gfx1250) — matrix operand order repaired and checked by reading, not compiled, not run on hardware
//
#include <hip/hip_runtime.h>

#define NATOMS 4096
#define NEDGES 65536
#define NTRI   131072
#define NMOLS  64
#define NRBF   6
#define EMBED  128
#define NBIL   8
#define NCONV  4
#define LSPH   7
#define NSPH   6
#define SBDIM  42
#define PI_D   3.14159265358979323846

typedef __bf16 bf16_t;
typedef __attribute__((ext_vector_type(16))) __bf16 v16bf;
typedef __attribute__((ext_vector_type(8)))  float  v8f;

enum { MODE_SILU = 0, MODE_SILU_RES = 1, MODE_SCALE_ACC = 2, MODE_LIN = 3 };

__device__ __forceinline__ bf16_t tobf(float f) {
  unsigned u = __float_as_uint(f);
  unsigned r = u + 0x7fffu + ((u >> 16) & 1u);          // round-to-nearest-even
  unsigned short h = (unsigned short)(r >> 16);
  bf16_t o; __builtin_memcpy(&o, &h, 2); return o;
}
__device__ __forceinline__ float silu_f(float x) { return x / (1.f + __expf(-x)); }

// ---------------------------------------------------------------------------
// Spherical Bessel zeros / norms / Y_l0 coefficients (fp64 bisection, 1 thread)
// ---------------------------------------------------------------------------
__device__ double jn_d(int l, double x) {
  double j0 = sin(x) / x;
  if (l == 0) return j0;
  double j1 = sin(x) / (x * x) - cos(x) / x;
  double jm = j0, jc = j1;
  for (int i = 2; i <= l; ++i) { double t = (2.0 * i - 1.0) / x * jc - jm; jm = jc; jc = t; }
  return jc;
}

__global__ void sph_init(float* sphZ, float* sphN, float* cbfc) {
  if (threadIdx.x != 0 || blockIdx.x != 0) return;
  double z[13];
  for (int k = 0; k < 13; ++k) z[k] = PI_D * (k + 1);
  for (int n = 0; n < NSPH; ++n) sphZ[n] = (float)z[n];
  int len = 13;
  for (int l = 1; l < LSPH; ++l) {
    double nz[13];
    for (int k = 0; k < len - 1; ++k) {
      double lo = z[k], hi = z[k + 1];
      double flo = jn_d(l, lo);
      for (int it = 0; it < 80; ++it) {
        double mid = 0.5 * (lo + hi);
        double fm = jn_d(l, mid);
        if (flo * fm <= 0.0) hi = mid; else { lo = mid; flo = fm; }
      }
      nz[k] = 0.5 * (lo + hi);
    }
    len -= 1;
    for (int k = 0; k < len; ++k) z[k] = nz[k];
    for (int n = 0; n < NSPH; ++n) sphZ[l * NSPH + n] = (float)z[n];
  }
  for (int l = 0; l < LSPH; ++l)
    for (int n = 0; n < NSPH; ++n)
      sphN[l * NSPH + n] = (float)(sqrt(2.0) / fabs(jn_d(l + 1, (double)sphZ[l * NSPH + n])));
  for (int l = 0; l < LSPH; ++l) cbfc[l] = (float)sqrt((2.0 * l + 1.0) / (4.0 * PI_D));
}

// ---------------------------------------------------------------------------
// Edge geometry: distance, envelope, radial basis, spherical-radial parts
// ---------------------------------------------------------------------------
__global__ void edge_geom(const float* __restrict__ xyz, const int* __restrict__ nbr,
                          const float* __restrict__ freq, const float* __restrict__ sphZ,
                          const float* __restrict__ sphN, float* __restrict__ e_rbf,
                          float* __restrict__ rad) {
  int e = blockIdx.x * blockDim.x + threadIdx.x;
  if (e >= NEDGES) return;
  int i = nbr[2 * e], j = nbr[2 * e + 1];
  float dx = xyz[3 * i] - xyz[3 * j];
  float dy = xyz[3 * i + 1] - xyz[3 * j + 1];
  float dz = xyz[3 * i + 2] - xyz[3 * j + 2];
  float d = sqrtf(dx * dx + dy * dy + dz * dz);
  float ds = d / 5.0f;
  float env = 0.f;
  if (ds < 1.f) {
    float d2 = ds * ds, d5 = d2 * d2 * ds;                      // p = 6
    env = 1.f / ds - 28.f * d5 + 48.f * d5 * ds - 21.f * d5 * d2;
  }
  for (int r = 0; r < NRBF; ++r) e_rbf[(size_t)e * NRBF + r] = env * sinf(freq[r] * ds);
  for (int l = 0; l < LSPH; ++l)
    for (int n = 0; n < NSPH; ++n) {
      float x = sphZ[l * NSPH + n] * ds;
      float j0 = sinf(x) / x;
      float jl = j0;
      if (l > 0) {
        float j1 = sinf(x) / (x * x) - cosf(x) / x;
        float jm = j0, jc = j1;
        for (int qq = 2; qq <= l; ++qq) { float t = (2.f * qq - 1.f) / x * jc - jm; jm = jc; jc = t; }
        jl = jc;
      }
      rad[(size_t)e * SBDIM + l * NSPH + n] = env * sphN[l * NSPH + n] * jl;
    }
}

// Triplet: angle -> Legendre -> a_sbf[T,42] = rad[kj] * cbf
__global__ void trip_sbf(const float* __restrict__ xyz, const int* __restrict__ al,
                         const int* __restrict__ kj, const float* __restrict__ rad,
                         const float* __restrict__ cbfc, float* __restrict__ asbf) {
  int t = blockIdx.x * blockDim.x + threadIdx.x;
  if (t >= NTRI) return;
  int a0 = al[3 * t], a1 = al[3 * t + 1], a2 = al[3 * t + 2];
  float jx = xyz[3 * a0] - xyz[3 * a1], jy = xyz[3 * a0 + 1] - xyz[3 * a1 + 1], jz = xyz[3 * a0 + 2] - xyz[3 * a1 + 2];
  float kx = xyz[3 * a2] - xyz[3 * a1], ky = xyz[3 * a2 + 1] - xyz[3 * a1 + 1], kz = xyz[3 * a2 + 2] - xyz[3 * a1 + 2];
  float dp = jx * kx + jy * ky + jz * kz;
  float cx1 = jy * kz - jz * ky, cy1 = jz * kx - jx * kz, cz1 = jx * ky - jy * kx;
  float yv = sqrtf(cx1 * cx1 + cy1 * cy1 + cz1 * cz1);
  float c = cosf(atan2f(yv, dp));
  float P[LSPH]; P[0] = 1.f; P[1] = c;
  for (int l = 2; l < LSPH; ++l) P[l] = ((2.f * l - 1.f) * c * P[l - 1] - (l - 1.f) * P[l - 2]) / (float)l;
  const float* rr = rad + (size_t)kj[t] * SBDIM;
  float* out = asbf + (size_t)t * SBDIM;
  for (int l = 0; l < LSPH; ++l) {
    float cb = cbfc[l] * P[l];
    for (int n = 0; n < NSPH; ++n) out[l * NSPH + n] = rr[l * NSPH + n] * cb;
  }
}

// sb[T,8] = a_sbf @ W_sbf   (K=42 too small for WMMA; VALU)
__global__ void sbf_proj(const float* __restrict__ asbf, const float* __restrict__ Wsbf,
                         float* __restrict__ sb) {
  int idx = blockIdx.x * blockDim.x + threadIdx.x;
  if (idx >= NTRI * NBIL) return;
  int t = idx >> 3, j = idx & 7;
  const float* a = asbf + (size_t)t * SBDIM;
  float s = 0.f;
  for (int k = 0; k < SBDIM; ++k) s += a[k] * Wsbf[k * NBIL + j];
  sb[idx] = s;
}

// silu(e_rbf @ W + b)  (K=6; VALU)
__global__ void rbf_dense_silu(const float* __restrict__ e_rbf, const float* __restrict__ W,
                               const float* __restrict__ b, float* __restrict__ out) {
  int idx = blockIdx.x * blockDim.x + threadIdx.x;
  if (idx >= NEDGES * EMBED) return;
  int e = idx >> 7, i = idx & 127;
  float a = b[i];
  for (int r = 0; r < NRBF; ++r) a += e_rbf[(size_t)e * NRBF + r] * W[r * EMBED + i];
  out[idx] = silu_f(a);
}

// cat = [h[nbr0] | h[nbr1] | e_act]  -> bf16 [E,384]
__global__ void build_cat(const float* __restrict__ emb, const int* __restrict__ z,
                          const int* __restrict__ nbr, const float* __restrict__ eact,
                          bf16_t* __restrict__ cat) {
  int idx = blockIdx.x * blockDim.x + threadIdx.x;
  if (idx >= NEDGES * 3 * EMBED) return;
  int e = idx / (3 * EMBED), k = idx % (3 * EMBED);
  float v;
  if (k < EMBED)            v = emb[(size_t)z[nbr[2 * e]] * EMBED + k];
  else if (k < 2 * EMBED)   v = emb[(size_t)z[nbr[2 * e + 1]] * EMBED + (k - EMBED)];
  else                      v = eact[(size_t)e * EMBED + (k - 2 * EMBED)];
  cat[idx] = tobf(v);
}

// h[e,i] *= (e_rbf[e,:] . W_rbf[:,i])
__global__ void apply_g(float* __restrict__ h, const float* __restrict__ e_rbf,
                        const float* __restrict__ Wrbf) {
  int idx = blockIdx.x * blockDim.x + threadIdx.x;
  if (idx >= NEDGES * EMBED) return;
  int e = idx >> 7, i = idx & 127;
  float g = 0.f;
  for (int r = 0; r < NRBF; ++r) g += e_rbf[(size_t)e * NRBF + r] * Wrbf[r * EMBED + i];
  h[idx] *= g;
}

__global__ void gather_bf(const float* __restrict__ src, const int* __restrict__ gidx,
                          bf16_t* __restrict__ dst) {
  int idx = blockIdx.x * blockDim.x + threadIdx.x;
  if (idx >= NTRI * EMBED) return;
  int t = idx >> 7, i = idx & 127;
  dst[idx] = tobf(src[(size_t)gidx[t] * EMBED + i]);
}

__global__ void scatter_add_rows(const float* __restrict__ src, const int* __restrict__ gidx,
                                 float* __restrict__ dst) {
  int idx = blockIdx.x * blockDim.x + threadIdx.x;
  if (idx >= NTRI * EMBED) return;
  int t = idx >> 7, i = idx & 127;
  atomicAdd(&dst[(size_t)gidx[t] * EMBED + i], src[idx]);
}

// t_atoms += scatter( (e_rbf @ W_rbf) * m  by nbr0 )
__global__ void scatter_out(const float* __restrict__ e_rbf, const float* __restrict__ Wrbf,
                            const float* __restrict__ m, const int* __restrict__ nbr,
                            float* __restrict__ tat) {
  int idx = blockIdx.x * blockDim.x + threadIdx.x;
  if (idx >= NEDGES * EMBED) return;
  int e = idx >> 7, i = idx & 127;
  float g = 0.f;
  for (int r = 0; r < NRBF; ++r) g += e_rbf[(size_t)e * NRBF + r] * Wrbf[r * EMBED + i];
  atomicAdd(&tat[(size_t)nbr[2 * e] * EMBED + i], g * m[idx]);
}

__global__ void cvt_bf(const float* __restrict__ src, bf16_t* __restrict__ dst, int n) {
  int idx = blockIdx.x * blockDim.x + threadIdx.x;
  if (idx < n) dst[idx] = tobf(src[idx]);
}

__global__ void wout_acc(const float* __restrict__ t, const float* __restrict__ Wout,
                         float* __restrict__ atom_out) {
  int a = blockIdx.x * blockDim.x + threadIdx.x;
  if (a >= NATOMS) return;
  const float* row = t + (size_t)a * EMBED;
  float s = 0.f;
  for (int i = 0; i < EMBED; ++i) s += row[i] * Wout[i];
  atom_out[a] += s;
}

__global__ void mol_reduce(const float* __restrict__ atom_out, const int* __restrict__ mol,
                           float* __restrict__ out) {
  int a = blockIdx.x * blockDim.x + threadIdx.x;
  if (a >= NATOMS) return;
  atomicAdd(&out[mol[a]], atom_out[a]);
}

// ---------------------------------------------------------------------------
// B-operand pack: [K,128] (arbitrary strides) f32 -> bf16 [K/32][128][32]
// so each lane's 16 K-values are two 16-byte chunks, at k = 8h and k = 16 + 8h.
// ---------------------------------------------------------------------------
__global__ void pack_b(const float* __restrict__ src, bf16_t* __restrict__ dst,
                       int K, int sk, int sn, int offp) {
  int idx = blockIdx.x * blockDim.x + threadIdx.x;
  if (idx >= K * EMBED) return;
  int k = idx >> 7, n = idx & 127;
  float v = src[(size_t)k * sk + (size_t)n * sn + offp];
  dst[((size_t)((k >> 5) * EMBED + n)) * 32 + (k & 31)] = tobf(v);
}

// ---------------------------------------------------------------------------
// WMMA GEMM, N = 128 fixed.  8 waves/block arranged 4(M)x2(N); each wave owns
// a 32x32 output tile as 2x2 WMMA subtiles (4 v8f accumulators).  Per K=32
// step: 2 A frags + 2 B frags -> 4 v_wmma_f32_16x16x32_bf16, i.e. every
// fragment is used twice (16 FLOP/byte of L2 traffic vs 8 for 1x1 tiling).
// A: row-major bf16 [M,K].  Bp: packed as above.  Fused epilogues.
// ---------------------------------------------------------------------------
__global__ __launch_bounds__(256)
void wmma_gemm_n128(const bf16_t* __restrict__ A, const bf16_t* __restrict__ Bp,
                    int M, int K,
                    const float* __restrict__ bias, const float* __restrict__ resid,
                    const float* __restrict__ rowscale, int rs_stride,
                    float* __restrict__ Cf, bf16_t* __restrict__ Cb, int mode) {
  const int lane = threadIdx.x & 31;
  const int w = threadIdx.x >> 5;
  const int wm = w & 3, wn = w >> 2;
  const int m0 = blockIdx.x * 128 + wm * 32;            // wave M-origin (32 rows)
  const int n0 = blockIdx.y * 64 + wn * 32;             // wave N-origin (32 cols)
  const int half = lane >> 4;
  const int l16 = lane & 15;

  v8f acc00 = {0.f,0.f,0.f,0.f,0.f,0.f,0.f,0.f};
  v8f acc01 = acc00, acc10 = acc00, acc11 = acc00;

  const int ktn = K >> 5;
  // A 16-bit fragment layout: lanes 0-15 hold K {0..7,16..23}, lanes 16-31 {8..15,24..31}
  const bf16_t* aptr0 = A + (size_t)(m0 + l16) * K + half * 8;
  const bf16_t* aptr1 = A + (size_t)(m0 + 16 + l16) * K + half * 8;
  const bf16_t* bptr0 = Bp + (size_t)(n0 + l16) * 32 + half * 8;
  const bf16_t* bptr1 = Bp + (size_t)(n0 + 16 + l16) * 32 + half * 8;
  for (int kt = 0; kt < ktn; ++kt) {
    union { v16bf v; uint4 q[2]; } fa0, fa1, fb0, fb1;
    fa0.q[0] = *(const uint4*)(aptr0);
    fa0.q[1] = *(const uint4*)(aptr0 + 16);
    fa1.q[0] = *(const uint4*)(aptr1);
    fa1.q[1] = *(const uint4*)(aptr1 + 16);
    fb0.q[0] = *(const uint4*)(bptr0);
    fb0.q[1] = *(const uint4*)(bptr0 + 16);
    fb1.q[0] = *(const uint4*)(bptr1);
    fb1.q[1] = *(const uint4*)(bptr1 + 16);
    acc00 = __builtin_amdgcn_wmma_f32_16x16x32_bf16(false, fa0.v, false, fb0.v, (short)0, acc00, false, false);
    acc01 = __builtin_amdgcn_wmma_f32_16x16x32_bf16(false, fa0.v, false, fb1.v, (short)0, acc01, false, false);
    acc10 = __builtin_amdgcn_wmma_f32_16x16x32_bf16(false, fa1.v, false, fb0.v, (short)0, acc10, false, false);
    acc11 = __builtin_amdgcn_wmma_f32_16x16x32_bf16(false, fa1.v, false, fb1.v, (short)0, acc11, false, false);
    aptr0 += 32; aptr1 += 32;
    bptr0 += (size_t)EMBED * 32;
    bptr1 += (size_t)EMBED * 32;
  }

  v8f* accs[4] = { &acc00, &acc01, &acc10, &acc11 };
#pragma unroll
  for (int mt = 0; mt < 2; ++mt) {
#pragma unroll
    for (int nt = 0; nt < 2; ++nt) {
      v8f& acc = *accs[mt * 2 + nt];
      const int col = n0 + nt * 16 + l16;
      const float bval = (bias != nullptr) ? bias[col] : 0.f;
#pragma unroll
      for (int r = 0; r < 8; ++r) {
        int row = m0 + mt * 16 + r + half * 8;          // C/D layout: vgpr r -> M=r / r+8
        size_t idx = (size_t)row * EMBED + col;
        float v = acc[r];
        if (mode == MODE_SCALE_ACC) {
          Cf[idx] += rowscale[(size_t)row * rs_stride] * v;
        } else {
          v += bval;
          if (mode != MODE_LIN) v = silu_f(v);
          if (mode == MODE_SILU_RES) v += resid[idx];
          Cf[idx] = v;
          if (Cb) Cb[idx] = tobf(v);
        }
      }
    }
  }
}

// ---------------------------------------------------------------------------
extern "C" void kernel_launch(void* const* d_in, const int* in_sizes, int n_in,
                              void* d_out, int out_size, void* d_ws, size_t ws_size,
                              hipStream_t stream) {
  (void)in_sizes; (void)n_in; (void)out_size; (void)ws_size;
  const float* xyz = (const float*)d_in[0];
  const int*   zat = (const int*)d_in[1];
  const int*   nbr = (const int*)d_in[2];
  const int*   al  = (const int*)d_in[3];
  const int*   kj  = (const int*)d_in[4];
  const int*   ji  = (const int*)d_in[5];
  const int*   mol = (const int*)d_in[6];

  int q = 7;
  auto FP = [&](int i) { return (const float*)d_in[i]; };
  const float* emb_tab   = FP(q++);
  const float* emb_W     = FP(q++);
  const float* emb_b     = FP(q++);
  const float* emb_rbf_W = FP(q++);
  const float* emb_rbf_b = FP(q++);
  struct ResP { const float *W1, *b1, *W2, *b2; };
  struct IntP {
    const float *W_bil, *W_ji, *W_kj, *W_rbf, *W_sbf, *W_skip, *b_ji, *b_kj, *b_skip;
    ResP ra[2]; ResP rb;
  };
  IntP ip[NCONV];
  for (int c = 0; c < NCONV; ++c) {
    ip[c].W_bil = FP(q++); ip[c].W_ji = FP(q++); ip[c].W_kj = FP(q++);
    ip[c].W_rbf = FP(q++); ip[c].W_sbf = FP(q++); ip[c].W_skip = FP(q++);
    ip[c].b_ji = FP(q++); ip[c].b_kj = FP(q++); ip[c].b_skip = FP(q++);
    for (int r = 0; r < 2; ++r) { ip[c].ra[r].W1 = FP(q++); ip[c].ra[r].b1 = FP(q++);
                                  ip[c].ra[r].W2 = FP(q++); ip[c].ra[r].b2 = FP(q++); }
    ip[c].rb.W1 = FP(q++); ip[c].rb.b1 = FP(q++); ip[c].rb.W2 = FP(q++); ip[c].rb.b2 = FP(q++);
  }
  struct OutP { const float *W_out, *W_rbf, *dW[3], *db[3]; };
  OutP op[NCONV + 1];
  for (int o = 0; o <= NCONV; ++o) {
    op[o].W_out = FP(q++); op[o].W_rbf = FP(q++);
    for (int d = 0; d < 3; ++d) { op[o].dW[d] = FP(q++); op[o].db[d] = FP(q++); }
  }
  const float* rbf_freq = FP(q++);

  // ---- workspace bump allocator ----
  char* wsb = (char*)d_ws;
  size_t off = 0;
  auto alloc = [&](size_t bytes) -> void* {
    void* r = (void*)(wsb + off);
    off += (bytes + 255) & ~(size_t)255;
    return r;
  };
  float*  sphZ  = (float*)alloc(SBDIM * 4);
  float*  sphN  = (float*)alloc(SBDIM * 4);
  float*  cbfc  = (float*)alloc(LSPH * 4);
  float*  e_rbf = (float*)alloc((size_t)NEDGES * NRBF * 4);
  float*  rad   = (float*)alloc((size_t)NEDGES * SBDIM * 4);
  float*  asbf  = (float*)alloc((size_t)NTRI * SBDIM * 4);
  float*  sb    = (float*)alloc((size_t)NTRI * NBIL * 4);
  float*  mE    = (float*)alloc((size_t)NEDGES * EMBED * 4);   // m (f32)
  float*  bufG  = (float*)alloc((size_t)NEDGES * EMBED * 4);
  float*  bufH  = (float*)alloc((size_t)NEDGES * EMBED * 4);
  bf16_t* mF    = (bf16_t*)alloc((size_t)NEDGES * EMBED * 2);  // m (bf16)
  bf16_t* bufI  = (bf16_t*)alloc((size_t)NEDGES * EMBED * 2);
  bf16_t* bufF2 = (bf16_t*)alloc((size_t)NEDGES * EMBED * 2);
  bf16_t* xkjJ  = (bf16_t*)alloc((size_t)NTRI * EMBED * 2);
  float*  TT    = (float*)alloc((size_t)NTRI * EMBED * 4);     // also reused as cat bf16
  float*  t_at  = (float*)alloc((size_t)NATOMS * EMBED * 4);
  float*  ta2   = (float*)alloc((size_t)NATOMS * EMBED * 4);
  bf16_t* tbA   = (bf16_t*)alloc((size_t)NATOMS * EMBED * 2);
  bf16_t* tbB   = (bf16_t*)alloc((size_t)NATOMS * EMBED * 2);
  float*  atom_out = (float*)alloc((size_t)NATOMS * 4);

  auto packW = [&](const float* src, int K, int sk, int sn, int offp) -> bf16_t* {
    bf16_t* dst = (bf16_t*)alloc((size_t)K * EMBED * sizeof(bf16_t));
    int n = K * EMBED;
    pack_b<<<dim3((n + 255) / 256), dim3(256), 0, stream>>>(src, dst, K, sk, sn, offp);
    return dst;
  };
  auto gemm = [&](const bf16_t* A, const bf16_t* Bp, int M, int K, const float* bias,
                  const float* resid, const float* rowscale, int rs_stride,
                  float* Cf, bf16_t* Cb, int mode) {
    wmma_gemm_n128<<<dim3(M / 128, 2), dim3(256), 0, stream>>>(
        A, Bp, M, K, bias, resid, rowscale, rs_stride, Cf, Cb, mode);
  };
  auto G1 = [](int n) { return dim3((n + 255) / 256); };

  // ---- pack all WMMA B operands to bf16 fragment layout ----
  bf16_t* p_embW = packW(emb_W, 3 * EMBED, EMBED, 1, 0);
  bf16_t *pWji[NCONV], *pWkj[NCONV], *pWskip[NCONV];
  bf16_t *pRa[NCONV][2][2], *pRb[NCONV][2], *pBil[NCONV][NBIL];
  for (int c = 0; c < NCONV; ++c) {
    pWji[c]   = packW(ip[c].W_ji,   EMBED, EMBED, 1, 0);
    pWkj[c]   = packW(ip[c].W_kj,   EMBED, EMBED, 1, 0);
    pWskip[c] = packW(ip[c].W_skip, EMBED, EMBED, 1, 0);
    for (int r = 0; r < 2; ++r) {
      pRa[c][r][0] = packW(ip[c].ra[r].W1, EMBED, EMBED, 1, 0);
      pRa[c][r][1] = packW(ip[c].ra[r].W2, EMBED, EMBED, 1, 0);
    }
    pRb[c][0] = packW(ip[c].rb.W1, EMBED, EMBED, 1, 0);
    pRb[c][1] = packW(ip[c].rb.W2, EMBED, EMBED, 1, 0);
    // bilinear slice j:  B[k=l][n=i] = W_bil[i, j, l] = src[i*1024 + j*128 + l]
    for (int j = 0; j < NBIL; ++j)
      pBil[c][j] = packW(ip[c].W_bil, EMBED, 1, NBIL * EMBED, j * EMBED);
  }
  bf16_t* pD[NCONV + 1][3];
  for (int o = 0; o <= NCONV; ++o)
    for (int d = 0; d < 3; ++d) pD[o][d] = packW(op[o].dW[d], EMBED, EMBED, 1, 0);

  // ---- geometry & bases ----
  sph_init<<<1, 1, 0, stream>>>(sphZ, sphN, cbfc);
  edge_geom<<<G1(NEDGES), 256, 0, stream>>>(xyz, nbr, rbf_freq, sphZ, sphN, e_rbf, rad);
  trip_sbf<<<G1(NTRI), 256, 0, stream>>>(xyz, al, kj, rad, cbfc, asbf);

  // ---- embedding: m = silu([h_i|h_j|e] @ emb_W + b) ----
  rbf_dense_silu<<<G1(NEDGES * EMBED), 256, 0, stream>>>(e_rbf, emb_rbf_W, emb_rbf_b, bufH);
  build_cat<<<G1(NEDGES * 3 * EMBED), 256, 0, stream>>>(emb_tab, zat, nbr, bufH, (bf16_t*)TT);
  gemm((const bf16_t*)TT, p_embW, NEDGES, 3 * EMBED, emb_b, nullptr, nullptr, 0,
       mE, mF, MODE_SILU);

  hipMemsetAsync(atom_out, 0, (size_t)NATOMS * 4, stream);
  hipMemsetAsync(d_out, 0, (size_t)NMOLS * 4, stream);

  auto out_block = [&](int o) {
    hipMemsetAsync(t_at, 0, (size_t)NATOMS * EMBED * 4, stream);
    scatter_out<<<G1(NEDGES * EMBED), 256, 0, stream>>>(e_rbf, op[o].W_rbf, mE, nbr, t_at);
    cvt_bf<<<G1(NATOMS * EMBED), 256, 0, stream>>>(t_at, tbA, NATOMS * EMBED);
    gemm(tbA, pD[o][0], NATOMS, EMBED, op[o].db[0], nullptr, nullptr, 0, ta2, tbB, MODE_SILU);
    gemm(tbB, pD[o][1], NATOMS, EMBED, op[o].db[1], nullptr, nullptr, 0, ta2, tbA, MODE_SILU);
    gemm(tbA, pD[o][2], NATOMS, EMBED, op[o].db[2], nullptr, nullptr, 0, ta2, nullptr, MODE_SILU);
    wout_acc<<<G1(NATOMS), 256, 0, stream>>>(ta2, op[o].W_out, atom_out);
  };
  out_block(0);

  // ---- interaction blocks ----
  for (int c = 0; c < NCONV; ++c) {
    sbf_proj<<<G1(NTRI * NBIL), 256, 0, stream>>>(asbf, ip[c].W_sbf, sb);
    // x_ji -> G ; x_kj -> H
    gemm(mF, pWji[c], NEDGES, EMBED, ip[c].b_ji, nullptr, nullptr, 0, bufG, nullptr, MODE_SILU);
    gemm(mF, pWkj[c], NEDGES, EMBED, ip[c].b_kj, nullptr, nullptr, 0, bufH, nullptr, MODE_SILU);
    apply_g<<<G1(NEDGES * EMBED), 256, 0, stream>>>(bufH, e_rbf, ip[c].W_rbf);
    gather_bf<<<G1(NTRI * EMBED), 256, 0, stream>>>(bufH, kj, xkjJ);
    // bilinear: TT += sb[:,j] * (x_kj @ W_bil[:,j,:]^T), then scatter to edges
    hipMemsetAsync(TT, 0, (size_t)NTRI * EMBED * 4, stream);
    for (int j = 0; j < NBIL; ++j)
      gemm(xkjJ, pBil[c][j], NTRI, EMBED, nullptr, nullptr, sb + j, NBIL, TT, nullptr,
           MODE_SCALE_ACC);
    scatter_add_rows<<<G1(NTRI * EMBED), 256, 0, stream>>>(TT, ji, bufG);
    // res_before: x = G -> H
    cvt_bf<<<G1(NEDGES * EMBED), 256, 0, stream>>>(bufG, bufI, NEDGES * EMBED);
    gemm(bufI, pRb[c][0], NEDGES, EMBED, ip[c].rb.b1, nullptr, nullptr, 0, bufH, bufF2, MODE_SILU);
    gemm(bufF2, pRb[c][1], NEDGES, EMBED, ip[c].rb.b2, bufG, nullptr, 0, bufH, bufI, MODE_SILU_RES);
    // skip: x = silu(x@W_skip+b) + m   -> G
    gemm(bufI, pWskip[c], NEDGES, EMBED, ip[c].b_skip, mE, nullptr, 0, bufG, bufF2, MODE_SILU_RES);
    // res_after[0]: x = G -> H
    gemm(bufF2, pRa[c][0][0], NEDGES, EMBED, ip[c].ra[0].b1, nullptr, nullptr, 0, bufH, bufI, MODE_SILU);
    gemm(bufI,  pRa[c][0][1], NEDGES, EMBED, ip[c].ra[0].b2, bufG, nullptr, 0, bufH, bufF2, MODE_SILU_RES);
    // res_after[1]: x = H -> E (new m)
    gemm(bufF2, pRa[c][1][0], NEDGES, EMBED, ip[c].ra[1].b1, nullptr, nullptr, 0, mE, bufI, MODE_SILU);
    gemm(bufI,  pRa[c][1][1], NEDGES, EMBED, ip[c].ra[1].b2, bufH, nullptr, 0, mE, mF, MODE_SILU_RES);
    out_block(c + 1);
  }

  mol_reduce<<<G1(NATOMS), 256, 0, stream>>>(atom_out, mol, (float*)d_out);
}
